// ResnetBlocWithAttn_69939247448244
// MI455X (gfx1250) — matrix operand order repaired and checked by reading, not compiled, not run on hardware
//
#include <hip/hip_runtime.h>
#include <hip/hip_bf16.h>

typedef _Float16 f16;
typedef __attribute__((ext_vector_type(16))) _Float16 v16h;
typedef __attribute__((ext_vector_type(8)))  _Float16 v8h;
typedef __attribute__((ext_vector_type(8)))  float    v8f;

#define Bn 4
#define Cc 128
#define Hh 64
#define Ww 64
#define HW 4096
#define Ee 512
#define Gg 32
#define CPG 4           // channels per group
#define QSCALE 0.08838834764831843f   // 1/sqrt(128)

// Set to 0 if the assembler rejects the async-to-LDS mnemonic.
#define ASYNC_COPY 1

// ---------------- WMMA helpers (gfx1250 wave32 layouts, ISA 7.12.2) ----------
__device__ inline v8f wmma_f16(v16h a, v16h b, v8f c) {
  return __builtin_amdgcn_wmma_f32_16x16x32_f16(
      false, a, false, b, (short)0, c, false, false);
}

// A matrix 16x32 f16: row m = lane&15; lane<16 holds k {0..7,16..23}, lane>=16 k {8..15,24..31}.
// rowbase points at &A[m][0] with k-stride 1.
__device__ inline v16h load_A16x32(const f16* rowbase, int lane) {
  int h8 = (lane & 16) ? 8 : 0;
  v8h x0 = *(const v8h*)(rowbase + h8);
  v8h x1 = *(const v8h*)(rowbase + 16 + h8);
  v16h r;
#pragma unroll
  for (int i = 0; i < 8; ++i) { r[i] = x0[i]; r[i + 8] = x1[i]; }
  return r;
}

// B matrix 32x16 f16: col n = lane&15; lane<16 holds k {0..7,16..23}, lane>=16 k {8..15,24..31}.
// colbase points at &B[0][n] with k-stride 1 (k contiguous in memory).
__device__ inline v16h load_B32x16(const f16* colbase, int lane) {
  int h8 = (lane & 16) ? 8 : 0;
  v8h x0 = *(const v8h*)(colbase + h8);
  v8h x1 = *(const v8h*)(colbase + 16 + h8);
  v16h r;
#pragma unroll
  for (int i = 0; i < 8; ++i) { r[i] = x0[i]; r[i + 8] = x1[i]; }
  return r;
}
// C/D f32 16x16: n = lane&15, reg r -> m = r + ((lane&16)?8:0)

// ---------------- async global -> LDS (CDNA5 ASYNCcnt path) ----------------
__device__ inline void async_g2l_b128(const f16* gsrc, f16* ldst) {
#if ASYNC_COPY
  unsigned long long g = (unsigned long long)gsrc;
  unsigned l = (unsigned)(unsigned long long)ldst;   // low 32 bits = LDS offset
  asm volatile("global_load_async_to_lds_b128 %0, %1, off"
               :: "v"(l), "v"(g) : "memory");
#else
  *(v8h*)ldst = *(const v8h*)gsrc;
#endif
}
__device__ inline void async_wait_all() {
#if ASYNC_COPY
  asm volatile("s_wait_asynccnt 0" ::: "memory");
#endif
}

// ---------------- weight conversion ----------------
__global__ void conv_w_tap_kernel(const float* __restrict__ w, f16* __restrict__ wt) {
  int i = blockIdx.x * 256 + threadIdx.x;
  if (i >= 9 * Cc * Cc) return;
  int tap = i / (Cc * Cc);
  int rem = i % (Cc * Cc);
  int co = rem / Cc, ci = rem % Cc;
  wt[i] = (f16)w[((size_t)co * Cc + ci) * 9 + tap];   // OIHW -> [tap][co][ci]
}

__global__ void convert_f16_kernel(const float* __restrict__ s, f16* __restrict__ d, int n) {
  int i = blockIdx.x * 256 + threadIdx.x;
  if (i < n) d[i] = (f16)s[i];
}

// ---------------- time-embedding MLP: t = swish(emb) @ t_w + t_b -------------
__global__ void time_mlp_kernel(const float* __restrict__ emb, const float* __restrict__ tw,
                                const float* __restrict__ tb, float* __restrict__ t) {
  __shared__ float se[Ee];
  int b = blockIdx.x;
  for (int e = threadIdx.x; e < Ee; e += blockDim.x) {
    float v = emb[b * Ee + e];
    se[e] = v / (1.f + __expf(-v));
  }
  __syncthreads();
  int c = threadIdx.x;   // 128 threads
  float s = tb[c];
  for (int e = 0; e < Ee; ++e) s += se[e] * tw[e * Cc + c];
  t[b * Cc + c] = s;
}

// ---------------- GroupNorm ----------------
// layout: 0 = NCHW f32 source, 1 = NHWC f32 source
__global__ void gn_stats_kernel(const float* __restrict__ src, float* __restrict__ mean,
                                float* __restrict__ rstd, int layout) {
  __shared__ float ss[256], sq[256];
  int bg = blockIdx.x;
  int b = bg / Gg, g = bg % Gg;
  float s = 0.f, q = 0.f;
  for (int i = threadIdx.x; i < CPG * HW; i += 256) {
    int c, p;
    float v;
    if (layout == 0) {
      c = g * CPG + i / HW; p = i % HW;
      v = src[((size_t)b * Cc + c) * HW + p];
    } else {
      p = i / CPG; c = g * CPG + (i % CPG);
      v = src[((size_t)b * HW + p) * Cc + c];
    }
    s += v; q += v * v;
  }
  ss[threadIdx.x] = s; sq[threadIdx.x] = q;
  __syncthreads();
  for (int o = 128; o > 0; o >>= 1) {
    if (threadIdx.x < o) { ss[threadIdx.x] += ss[threadIdx.x + o]; sq[threadIdx.x] += sq[threadIdx.x + o]; }
    __syncthreads();
  }
  if (threadIdx.x == 0) {
    float m = ss[0] / (float)(CPG * HW);
    float var = sq[0] / (float)(CPG * HW) - m * m;
    mean[bg] = m;
    rstd[bg] = rsqrtf(var + 1e-5f);
  }
}

// normalize (+optional swish), write NHWC f16
__global__ void gn_apply_kernel(const float* __restrict__ src, const float* __restrict__ mean,
                                const float* __restrict__ rstd, const float* __restrict__ scale,
                                const float* __restrict__ bias, f16* __restrict__ dst,
                                int layout, int do_swish) {
  size_t i = (size_t)blockIdx.x * 256 + threadIdx.x;       // NHWC dst index
  if (i >= (size_t)Bn * HW * Cc) return;
  int c = (int)(i % Cc);
  size_t t = i / Cc;
  int p = (int)(t % HW);
  int b = (int)(t / HW);
  float v = (layout == 0) ? src[((size_t)b * Cc + c) * HW + p] : src[i];
  int bg = b * Gg + (c >> 2);
  v = (v - mean[bg]) * rstd[bg] * scale[c] + bias[c];
  if (do_swish) v = v / (1.f + __expf(-v));
  dst[i] = (f16)v;
}

// ---------------- 3x3 conv as implicit GEMM (WMMA) ----------------
// act: NHWC f16 [B][H][W][C]; wt: [9][co][ci] f16; bias: [C]
// mode 0: dst NHWC f32 h, extra = time vec [B][C]
// mode 1: dst NCHW f32 x2, extra = x NCHW f32 (residual)
__global__ __launch_bounds__(256)
void conv3x3_wmma_kernel(const f16* __restrict__ act, const f16* __restrict__ wt,
                         const float* __restrict__ bias, const float* __restrict__ extra,
                         float* __restrict__ dst, int mode) {
  int lane = threadIdx.x & 31, wave = threadIdx.x >> 5;
  int half = (lane >> 4) & 1, nl = lane & 15;
  int y = blockIdx.x % Hh, b = blockIdx.x / Hh;
  int co0 = wave * 16;
  v8f acc[4];
#pragma unroll
  for (int i = 0; i < 4; ++i) acc[i] = {};
  for (int tap = 0; tap < 9; ++tap) {
    int dy = tap / 3 - 1, dx = tap % 3 - 1;
    int yy = y + dy;
    if (yy < 0 || yy >= Hh) continue;             // uniform across block
    const f16* wrow = wt + (size_t)tap * Cc * Cc + (size_t)(co0 + nl) * Cc;
    const f16* arow = act + ((size_t)b * Hh + yy) * Ww * Cc;
#pragma unroll
    for (int ks = 0; ks < 4; ++ks) {
      v16h A = load_A16x32(wrow + ks * 32, lane);
#pragma unroll
      for (int nt = 0; nt < 4; ++nt) {
        int x = nt * 16 + nl + dx;
        v16h Bv = {};
        if (x >= 0 && x < Ww) Bv = load_B32x16(arow + (size_t)x * Cc + ks * 32, lane);
        acc[nt] = wmma_f16(A, Bv, acc[nt]);
      }
    }
  }
#pragma unroll
  for (int nt = 0; nt < 4; ++nt)
#pragma unroll
    for (int r = 0; r < 8; ++r) {
      int co = co0 + r + half * 8;
      int p = y * Ww + nt * 16 + nl;
      float v = acc[nt][r] + bias[co];
      if (mode == 0) {
        v += extra[b * Cc + co];
        dst[((size_t)b * HW + p) * Cc + co] = v;          // h NHWC
      } else {
        size_t idx = ((size_t)b * Cc + co) * HW + p;
        dst[idx] = v + extra[idx];                        // x2 NCHW
      }
    }
}

// ---------------- qkv 1x1 conv GEMM: [384x128] x [128 x HW] ----------------
// nact NHWC f16; writes qT[pix][c] (scaled), kT[pix][c], v[c][pix]
__global__ __launch_bounds__(256)
void qkv_gemm_kernel(const f16* __restrict__ nact, const f16* __restrict__ wq,
                     f16* __restrict__ qT, f16* __restrict__ kT, f16* __restrict__ vv) {
  int lane = threadIdx.x & 31, wave = threadIdx.x >> 5;
  int half = lane >> 4, nl = lane & 15;
  int b = blockIdx.x / (HW / 64);
  int pt0 = (blockIdx.x % (HW / 64)) * 64;
  for (int i = 0; i < 3; ++i) {
    int mt = wave + 8 * i;                       // 0..23 row tiles of 384
    const f16* wrow = wq + (size_t)(mt * 16 + nl) * Cc;
    v8f acc[4];
#pragma unroll
    for (int j = 0; j < 4; ++j) acc[j] = {};
#pragma unroll
    for (int ks = 0; ks < 4; ++ks) {
      v16h A = load_A16x32(wrow + ks * 32, lane);
#pragma unroll
      for (int nt = 0; nt < 4; ++nt) {
        int pix = pt0 + nt * 16 + nl;
        v16h Bv = load_B32x16(nact + ((size_t)b * HW + pix) * Cc + ks * 32, lane);
        acc[nt] = wmma_f16(A, Bv, acc[nt]);
      }
    }
#pragma unroll
    for (int nt = 0; nt < 4; ++nt)
#pragma unroll
      for (int r = 0; r < 8; ++r) {
        int row = mt * 16 + r + half * 8;
        int pix = pt0 + nt * 16 + nl;
        float val = acc[nt][r];
        if (row < Cc)
          qT[((size_t)b * HW + pix) * Cc + row] = (f16)(val * QSCALE);
        else if (row < 2 * Cc)
          kT[((size_t)b * HW + pix) * Cc + (row - Cc)] = (f16)val;
        else
          vv[((size_t)b * Cc + (row - 2 * Cc)) * HW + pix] = (f16)val;
      }
  }
}

// ---------------- flash attention: 4 waves (64 queries) per block ----------
// K/V j-tiles staged once per block into LDS via async global->LDS copies,
// shared by all 4 waves (4x reduction in K/V global traffic).
__global__ __launch_bounds__(128)
void flash_attn_kernel(const f16* __restrict__ qT, const f16* __restrict__ kT,
                       const f16* __restrict__ vv, f16* __restrict__ oT) {
  __shared__ __align__(32) f16 Ks[32 * 128];     // Ks[jl][c]  (8 KB)
  __shared__ __align__(32) f16 Vs[128 * 32];     // Vs[c][jl]  (8 KB)
  __shared__ __align__(32) f16 psh[4][16 * 32];  // per-wave P transpose (4 KB)
  int tid = threadIdx.x;
  int lane = tid & 31, wave = tid >> 5;
  int half = lane >> 4, nl = lane & 15;
  int b = blockIdx.x / (HW / 64);
  int i0 = (blockIdx.x % (HW / 64)) * 64 + wave * 16;   // this wave's 16 queries

  v16h Aq[4];
#pragma unroll
  for (int ks = 0; ks < 4; ++ks)
    Aq[ks] = load_A16x32(qT + ((size_t)b * HW + i0 + nl) * Cc + ks * 32, lane);
  v8f o[8];
#pragma unroll
  for (int i = 0; i < 8; ++i) o[i] = {};
  float mrow[8], lrow[8];
#pragma unroll
  for (int r = 0; r < 8; ++r) { mrow[r] = -3.0e38f; lrow[r] = 0.f; }

  for (int jb = 0; jb < HW; jb += 32) {
    __syncthreads();                             // WAR: previous tile fully consumed
    // stage K tile: 32 rows x 256B = 512 x 16B chunks
#pragma unroll
    for (int k = 0; k < 4; ++k) {
      int cid = tid + 128 * k;
      int jl = cid >> 4, seg = cid & 15;
      async_g2l_b128(kT + ((size_t)b * HW + jb + jl) * Cc + seg * 8,
                     Ks + jl * 128 + seg * 8);
    }
    // stage V tile: 128 rows x 64B = 512 x 16B chunks
#pragma unroll
    for (int k = 0; k < 4; ++k) {
      int cid = tid + 128 * k;
      int c = cid >> 2, seg = cid & 3;
      async_g2l_b128(vv + ((size_t)b * Cc + c) * HW + jb + seg * 8,
                     Vs + c * 32 + seg * 8);
    }
    async_wait_all();
    __syncthreads();                             // tiles visible to all waves

    v8f s0 = {}, s1 = {};
#pragma unroll
    for (int ks = 0; ks < 4; ++ks) {
      v16h B0 = load_B32x16(Ks + nl * 128 + ks * 32, lane);
      v16h B1 = load_B32x16(Ks + (16 + nl) * 128 + ks * 32, lane);
      s0 = wmma_f16(Aq[ks], B0, s0);
      s1 = wmma_f16(Aq[ks], B1, s1);
    }
    float alpha[8];
#pragma unroll
    for (int r = 0; r < 8; ++r) {
      float t = fmaxf(s0[r], s1[r]);             // butterfly stays within 16-lane half
      t = fmaxf(t, __shfl_xor(t, 1, 32));
      t = fmaxf(t, __shfl_xor(t, 2, 32));
      t = fmaxf(t, __shfl_xor(t, 4, 32));
      t = fmaxf(t, __shfl_xor(t, 8, 32));
      float mnew = fmaxf(mrow[r], t);
      alpha[r] = __expf(mrow[r] - mnew);
      mrow[r] = mnew;
      float p0 = __expf(s0[r] - mnew);
      float p1 = __expf(s1[r] - mnew);
      s0[r] = p0; s1[r] = p1;
      float rs = p0 + p1;
      rs += __shfl_xor(rs, 1, 32);
      rs += __shfl_xor(rs, 2, 32);
      rs += __shfl_xor(rs, 4, 32);
      rs += __shfl_xor(rs, 8, 32);
      lrow[r] = lrow[r] * alpha[r] + rs;
    }
#pragma unroll
    for (int cs = 0; cs < 8; ++cs)
#pragma unroll
      for (int r = 0; r < 8; ++r) o[cs][r] *= alpha[r];
    // D-layout P -> per-wave LDS [16 q][32 j] -> reload in A-layout
    f16* pw = &psh[wave][0];
#pragma unroll
    for (int r = 0; r < 8; ++r) {
      int m = r + half * 8;
      pw[m * 32 + nl] = (f16)s0[r];
      pw[m * 32 + 16 + nl] = (f16)s1[r];
    }
    asm volatile("s_wait_dscnt 0" ::: "memory"); // wave-private: DS in-order + wait
    v16h Ap = load_A16x32(pw + nl * 32, lane);
#pragma unroll
    for (int cs = 0; cs < 8; ++cs) {
      v16h Bv = load_B32x16(Vs + (cs * 16 + nl) * 32, lane);
      o[cs] = wmma_f16(Ap, Bv, o[cs]);
    }
  }
#pragma unroll
  for (int cs = 0; cs < 8; ++cs)
#pragma unroll
    for (int r = 0; r < 8; ++r) {
      int m = r + half * 8;
      oT[((size_t)b * HW + i0 + m) * Cc + cs * 16 + nl] = (f16)(o[cs][r] / lrow[r]);
    }
}

// ---------------- out 1x1 conv GEMM + bias + residual ----------------
__global__ __launch_bounds__(256)
void out_gemm_kernel(const f16* __restrict__ oT, const f16* __restrict__ wo,
                     const float* __restrict__ ob, const float* __restrict__ x2,
                     float* __restrict__ out) {
  int lane = threadIdx.x & 31, wave = threadIdx.x >> 5;
  int half = lane >> 4, nl = lane & 15;
  int b = blockIdx.x / (HW / 64);
  int pt0 = (blockIdx.x % (HW / 64)) * 64;
  int co0 = wave * 16;
  const f16* wrow = wo + (size_t)(co0 + nl) * Cc;
  v8f acc[4];
#pragma unroll
  for (int i = 0; i < 4; ++i) acc[i] = {};
#pragma unroll
  for (int ks = 0; ks < 4; ++ks) {
    v16h A = load_A16x32(wrow + ks * 32, lane);
#pragma unroll
    for (int nt = 0; nt < 4; ++nt) {
      int pix = pt0 + nt * 16 + nl;
      v16h Bv = load_B32x16(oT + ((size_t)b * HW + pix) * Cc + ks * 32, lane);
      acc[nt] = wmma_f16(A, Bv, acc[nt]);
    }
  }
#pragma unroll
  for (int nt = 0; nt < 4; ++nt)
#pragma unroll
    for (int r = 0; r < 8; ++r) {
      int co = co0 + r + half * 8;
      int pix = pt0 + nt * 16 + nl;
      size_t idx = ((size_t)b * Cc + co) * HW + pix;
      out[idx] = acc[nt][r] + ob[co] + x2[idx];
    }
}

// ---------------- launch ----------------
extern "C" void kernel_launch(void* const* d_in, const int* in_sizes, int n_in,
                              void* d_out, int out_size, void* d_ws, size_t ws_size,
                              hipStream_t stream) {
  const float* x       = (const float*)d_in[0];
  const float* temb    = (const float*)d_in[1];
  const float* gn1_s   = (const float*)d_in[2];
  const float* gn1_b   = (const float*)d_in[3];
  const float* conv1_w = (const float*)d_in[4];
  const float* conv1_b = (const float*)d_in[5];
  const float* t_w     = (const float*)d_in[6];
  const float* t_b     = (const float*)d_in[7];
  const float* gn2_s   = (const float*)d_in[8];
  const float* gn2_b   = (const float*)d_in[9];
  const float* conv2_w = (const float*)d_in[10];
  const float* conv2_b = (const float*)d_in[11];
  const float* agn_s   = (const float*)d_in[12];
  const float* agn_b   = (const float*)d_in[13];
  const float* qkv_w   = (const float*)d_in[14];
  const float* out_w   = (const float*)d_in[15];
  const float* out_b   = (const float*)d_in[16];
  float* out = (float*)d_out;

  char* p = (char*)d_ws;
  auto carve = [&](size_t bytes) -> void* {
    void* r = (void*)p;
    p += (bytes + 255) & ~(size_t)255;
    return r;
  };
  float* tvec  = (float*)carve(Bn * Cc * sizeof(float));
  f16* wt1     = (f16*)carve((size_t)9 * Cc * Cc * sizeof(f16));
  f16* wt2     = (f16*)carve((size_t)9 * Cc * Cc * sizeof(f16));
  f16* wqkv    = (f16*)carve((size_t)3 * Cc * Cc * sizeof(f16));
  f16* wout    = (f16*)carve((size_t)Cc * Cc * sizeof(f16));
  float* mean  = (float*)carve(Bn * Gg * sizeof(float));
  float* rstd  = (float*)carve(Bn * Gg * sizeof(float));
  f16* act     = (f16*)carve((size_t)Bn * HW * Cc * sizeof(f16));   // NHWC activations
  float* hbuf  = (float*)carve((size_t)Bn * Cc * HW * sizeof(float)); // h NHWC, reused as x2 NCHW
  f16* qT      = (f16*)carve((size_t)Bn * HW * Cc * sizeof(f16));
  f16* kT      = (f16*)carve((size_t)Bn * HW * Cc * sizeof(f16));
  f16* vbuf    = (f16*)carve((size_t)Bn * Cc * HW * sizeof(f16));
  f16* oT      = (f16*)carve((size_t)Bn * HW * Cc * sizeof(f16));

  const int elemN = Bn * HW * Cc;                 // 2097152

  // weight prep + time MLP (independent)
  conv_w_tap_kernel<<<(9 * Cc * Cc + 255) / 256, 256, 0, stream>>>(conv1_w, wt1);
  conv_w_tap_kernel<<<(9 * Cc * Cc + 255) / 256, 256, 0, stream>>>(conv2_w, wt2);
  convert_f16_kernel<<<(3 * Cc * Cc + 255) / 256, 256, 0, stream>>>(qkv_w, wqkv, 3 * Cc * Cc);
  convert_f16_kernel<<<(Cc * Cc + 255) / 256, 256, 0, stream>>>(out_w, wout, Cc * Cc);
  time_mlp_kernel<<<Bn, Cc, 0, stream>>>(temb, t_w, t_b, tvec);

  // GN1 + swish -> act (NHWC f16); conv1 (+bias+time) -> h (NHWC f32)
  gn_stats_kernel<<<Bn * Gg, 256, 0, stream>>>(x, mean, rstd, 0);
  gn_apply_kernel<<<(elemN + 255) / 256, 256, 0, stream>>>(x, mean, rstd, gn1_s, gn1_b, act, 0, 1);
  conv3x3_wmma_kernel<<<Bn * Hh, 256, 0, stream>>>(act, wt1, conv1_b, tvec, hbuf, 0);

  // GN2 + swish -> act; conv2 (+bias+residual x) -> x2 (NCHW f32, reuses hbuf)
  gn_stats_kernel<<<Bn * Gg, 256, 0, stream>>>(hbuf, mean, rstd, 1);
  gn_apply_kernel<<<(elemN + 255) / 256, 256, 0, stream>>>(hbuf, mean, rstd, gn2_s, gn2_b, act, 1, 1);
  conv3x3_wmma_kernel<<<Bn * Hh, 256, 0, stream>>>(act, wt2, conv2_b, x, hbuf, 1);
  float* x2 = hbuf;

  // attention: GN (no swish) -> act; qkv GEMM -> qT/kT/v; flash -> oT; out GEMM
  gn_stats_kernel<<<Bn * Gg, 256, 0, stream>>>(x2, mean, rstd, 0);
  gn_apply_kernel<<<(elemN + 255) / 256, 256, 0, stream>>>(x2, mean, rstd, agn_s, agn_b, act, 0, 0);
  qkv_gemm_kernel<<<Bn * (HW / 64), 256, 0, stream>>>(act, wqkv, qT, kT, vbuf);
  flash_attn_kernel<<<Bn * (HW / 64), 128, 0, stream>>>(qT, kT, vbuf, oT);
  out_gemm_kernel<<<Bn * (HW / 64), 256, 0, stream>>>(oT, wout, out_b, x2, out);
}
